// DeformableAttention_37503654428834
// MI455X (gfx1250) — matrix operand order repaired and checked by reading, not compiled, not run on hardware
//
#include <hip/hip_runtime.h>
#include <hip/hip_bf16.h>

typedef __attribute__((ext_vector_type(16))) __bf16 v16bf;
typedef __attribute__((ext_vector_type(8)))  __bf16 v8bf;
typedef __attribute__((ext_vector_type(8)))  float  v8f;

// ---------------------------------------------------------------------------
// fp32 -> bf16 conversion (for WMMA A/B operands)
// ---------------------------------------------------------------------------
__global__ __launch_bounds__(256)
void cvt_f32_bf16(const float* __restrict__ src, __bf16* __restrict__ dst, int n) {
  int i = blockIdx.x * blockDim.x + threadIdx.x;
  if (i < n) dst[i] = (__bf16)src[i];
}

// ---------------------------------------------------------------------------
// Y(R,O) = X(R,256) @ W(O,256)^T + bias, bf16 inputs, f32 accumulate.
// One wave -> 16x64 output tile via 4x v_wmma_f32_16x16x32_bf16 per K-step.
//
// wave32 fragment layouts as this code uses them (A/B K contiguous per lane half):
//   A (16x32 bf16): lane<16 holds row=lane, K 0..15; lane>=16 holds
//                   row=lane-16, K 16..31  -> two 16B loads per lane.
//   B (32x16 bf16): lane<16 holds col=lane, K 0..15; lane>=16 col=lane-16,
//                   K 16..31 -> one contiguous 32B load per lane (row of W).
//   D (16x16 f32) : vgpr r = row r (lanes 0..15) / row r+8 (lanes 16..31).
// ---------------------------------------------------------------------------
__global__ __launch_bounds__(256)
void gemm_bf16_wmma(const __bf16* __restrict__ X, const __bf16* __restrict__ Wt,
                    const float* __restrict__ bias, float* __restrict__ Y,
                    int R, int O) {
  const int K = 256;
  const int colGroups = O >> 6;             // 64 output cols per wave
  const int rowTiles  = (R + 15) >> 4;
  int wave = blockIdx.x * (blockDim.x >> 5) + (threadIdx.x >> 5);
  if (wave >= rowTiles * colGroups) return; // wave-uniform exit
  int rowTile = wave / colGroups;
  int cg      = wave - rowTile * colGroups;
  int r0 = rowTile << 4;
  int c0 = cg << 6;
  int lane = threadIdx.x & 31;
  int half = lane >> 4;
  int mrow = lane & 15;
  int row = r0 + mrow; if (row >= R) row = R - 1;   // clamp loads, guard stores

  const __bf16* xr = X + (size_t)row * K + half * 16;
  const __bf16* w0 = Wt + (size_t)(c0 +  0 + mrow) * K + half * 16;
  const __bf16* w1 = Wt + (size_t)(c0 + 16 + mrow) * K + half * 16;
  const __bf16* w2 = Wt + (size_t)(c0 + 32 + mrow) * K + half * 16;
  const __bf16* w3 = Wt + (size_t)(c0 + 48 + mrow) * K + half * 16;

  v8f zero = {};
  v8f acc[4]; acc[0] = zero; acc[1] = zero; acc[2] = zero; acc[3] = zero;

  #pragma unroll
  for (int k0 = 0; k0 < K; k0 += 32) {
    v8bf alo = *(const v8bf*)(xr + k0);        // K {k0..k0+7}   (+16 if half)
    v8bf ahi = *(const v8bf*)(xr + k0 + 8);    // K {k0+8..+15}  (+16 if half)
    v16bf a;
    #pragma unroll
    for (int i = 0; i < 8; ++i) { a[i] = alo[i]; a[8 + i] = ahi[i]; }
    v16bf b0 = *(const v16bf*)(w0 + k0);
    v16bf b1 = *(const v16bf*)(w1 + k0);
    v16bf b2 = *(const v16bf*)(w2 + k0);
    v16bf b3 = *(const v16bf*)(w3 + k0);
    acc[0] = __builtin_amdgcn_wmma_f32_16x16x32_bf16(false, a, false, b0, (short)0, acc[0], false, false);
    acc[1] = __builtin_amdgcn_wmma_f32_16x16x32_bf16(false, a, false, b1, (short)0, acc[1], false, false);
    acc[2] = __builtin_amdgcn_wmma_f32_16x16x32_bf16(false, a, false, b2, (short)0, acc[2], false, false);
    acc[3] = __builtin_amdgcn_wmma_f32_16x16x32_bf16(false, a, false, b3, (short)0, acc[3], false, false);
  }

  #pragma unroll
  for (int t = 0; t < 4; ++t) {
    int coln = c0 + (t << 4) + mrow;
    float bv = bias[coln];
    #pragma unroll
    for (int rr = 0; rr < 8; ++rr) {
      int orow = r0 + (half << 3) + rr;
      if (orow < R) Y[(size_t)orow * O + coln] = acc[t][rr] + bv;
    }
  }
}

// ---------------------------------------------------------------------------
// Fused softmax + multi-scale bilinear sampling + weighted sum.
// One wave per (n, q, head); lane = channel d (D=32 == wave32).
// v (fp32, 27 MB) lives in the 192 MB L2, so the 4 gather loads per sample
// are coalesced 128B L2 hits. Output written as out[n][q][m*32+d].
// ---------------------------------------------------------------------------
__global__ __launch_bounds__(256)
void deform_sample(const float* __restrict__ vf,   // (N,Lv,8,32) f32
                   const float* __restrict__ offf, // (N,Lq,8,4,4,2) f32
                   const float* __restrict__ awf,  // (N,Lq,8,16) raw logits
                   const float* __restrict__ ref,  // (N,Lq,4,4)
                   float* __restrict__ out,        // (N,Lq,256)
                   int N, int Lq) {
  const int Lv = 13294;
  int wid = blockIdx.x * (blockDim.x >> 5) + (threadIdx.x >> 5);
  int total = N * Lq * 8;
  if (wid >= total) return;
  int m = wid & 7;
  int q = (wid >> 3) % Lq;
  int n = wid / (Lq << 3);
  int d = threadIdx.x & 31;
  size_t nq = (size_t)n * Lq + q;

  // softmax over the 16 (level,point) logits
  const float* awp = awf + (nq * 8 + m) * 16;
  float w[16];
  float mx = -1e30f;
  #pragma unroll
  for (int i = 0; i < 16; ++i) { w[i] = awp[i]; mx = fmaxf(mx, w[i]); }
  float s = 0.f;
  #pragma unroll
  for (int i = 0; i < 16; ++i) { w[i] = __expf(w[i] - mx); s += w[i]; }
  float inv = 1.f / s;

  const float* op = offf + nq * 256 + m * 32;
  const float* rp = ref + nq * 16;
  const float* vb = vf + (size_t)n * Lv * 256 + m * 32 + d;

  const int   HS[4]   = {100, 50, 25, 13};
  const int   WS[4]   = {100, 50, 25, 13};
  const int   OFFS[4] = {0, 10000, 12500, 13125};

  float acc = 0.f;
  #pragma unroll
  for (int l = 0; l < 4; ++l) {
    const int Wl = WS[l], Hl = HS[l];
    const float Wf = (float)Wl, Hf = (float)Hl;
    const float* vbl = vb + (size_t)OFFS[l] * 256;
    float rx = rp[l * 4 + 0], ry = rp[l * 4 + 1];
    float rw = rp[l * 4 + 2], rh = rp[l * 4 + 3];
    #pragma unroll
    for (int p = 0; p < 4; ++p) {
      float ox = op[(l * 4 + p) * 2 + 0];
      float oy = op[(l * 4 + p) * 2 + 1];
      // loc = ref_xy + off/P * ref_wh * 0.5 ; x = loc.x*W - 0.5 (align_corners=F)
      float lx = rx + ox * 0.125f * rw;
      float ly = ry + oy * 0.125f * rh;
      float x = lx * Wf - 0.5f;
      float y = ly * Hf - 0.5f;
      float x0f = floorf(x), y0f = floorf(y);
      float fx = x - x0f, fy = y - y0f;
      int x0 = (int)x0f, y0 = (int)y0f;
      float wt = w[l * 4 + p];
      float w00 = (1.f - fx) * (1.f - fy) * wt;
      float w10 = fx * (1.f - fy) * wt;
      float w01 = (1.f - fx) * fy * wt;
      float w11 = fx * fy * wt;
      bool xi0 = (x0 >= 0) && (x0 < Wl);
      bool xi1 = (x0 + 1 >= 0) && (x0 + 1 < Wl);
      bool yi0 = (y0 >= 0) && (y0 < Hl);
      bool yi1 = (y0 + 1 >= 0) && (y0 + 1 < Hl);
      if (xi0 && yi0) acc += vbl[(size_t)(y0 * Wl + x0) * 256] * w00;
      if (xi1 && yi0) acc += vbl[(size_t)(y0 * Wl + x0 + 1) * 256] * w10;
      if (xi0 && yi1) acc += vbl[(size_t)((y0 + 1) * Wl + x0) * 256] * w01;
      if (xi1 && yi1) acc += vbl[(size_t)((y0 + 1) * Wl + x0 + 1) * 256] * w11;
    }
  }
  out[nq * 256 + m * 32 + d] = acc * inv;
}

// ---------------------------------------------------------------------------
extern "C" void kernel_launch(void* const* d_in, const int* in_sizes, int n_in,
                              void* d_out, int out_size, void* d_ws, size_t ws_size,
                              hipStream_t stream) {
  const float* query  = (const float*)d_in[0];
  const float* refpts = (const float*)d_in[1];
  const float* value  = (const float*)d_in[2];
  const float* W_val  = (const float*)d_in[3];
  const float* b_val  = (const float*)d_in[4];
  const float* W_off  = (const float*)d_in[5];
  const float* b_off  = (const float*)d_in[6];
  const float* W_attn = (const float*)d_in[7];
  const float* b_attn = (const float*)d_in[8];
  float* out = (float*)d_out;

  const int N = 2, Lq = 13294, C = 256;
  const int R = N * Lq;                  // 26588 rows for every GEMM
  const size_t nQ  = (size_t)R * C;      // query/value element count
  const size_t nWv = 256 * 256;
  const size_t nWa = 128 * 256;

  // workspace partition (hipMalloc base is 256B aligned; all chunks padded)
  char* ws = (char*)d_ws;
  size_t cur = 0;
  auto carve = [&](size_t bytes) -> char* {
    char* p = ws + cur;
    cur += (bytes + 255) & ~(size_t)255;
    return p;
  };
  __bf16* qb  = (__bf16*)carve(nQ * 2);        // query bf16
  __bf16* vb  = (__bf16*)carve(nQ * 2);        // value bf16
  __bf16* Wvb = (__bf16*)carve(nWv * 2);       // W_val bf16
  __bf16* Wob = (__bf16*)carve(nWv * 2);       // W_off bf16
  __bf16* Wab = (__bf16*)carve(nWa * 2);       // W_attn bf16
  float*  vf  = (float*)carve(nQ * 4);         // projected values (f32, L2-resident)
  float*  of  = (float*)carve(nQ * 4);         // sampling offsets
  float*  awf = (float*)carve((size_t)R * 128 * 4); // attention logits

  // 1) bf16 conversions
  cvt_f32_bf16<<<(int)((nQ + 255) / 256), 256, 0, stream>>>(query, qb, (int)nQ);
  cvt_f32_bf16<<<(int)((nQ + 255) / 256), 256, 0, stream>>>(value, vb, (int)nQ);
  cvt_f32_bf16<<<(int)((nWv + 255) / 256), 256, 0, stream>>>(W_val, Wvb, (int)nWv);
  cvt_f32_bf16<<<(int)((nWv + 255) / 256), 256, 0, stream>>>(W_off, Wob, (int)nWv);
  cvt_f32_bf16<<<(int)((nWa + 255) / 256), 256, 0, stream>>>(W_attn, Wab, (int)nWa);

  // 2) WMMA GEMMs: 8 waves/block, each wave a 16x64 tile
  int rowTiles = (R + 15) >> 4;                // 1662
  {
    int waves = rowTiles * (256 >> 6);
    int blocks = (waves + 7) / 8;
    gemm_bf16_wmma<<<blocks, 256, 0, stream>>>(vb, Wvb, b_val, vf, R, 256);
    gemm_bf16_wmma<<<blocks, 256, 0, stream>>>(qb, Wob, b_off, of, R, 256);
  }
  {
    int waves = rowTiles * (128 >> 6);
    int blocks = (waves + 7) / 8;
    gemm_bf16_wmma<<<blocks, 256, 0, stream>>>(qb, Wab, b_attn, awf, R, 128);
  }

  // 3) fused softmax + bilinear sampling + head combine
  {
    int waves = R * 8;                         // one wave per (n,q,head)
    int blocks = (waves + 7) / 8;              // 26588 blocks
    deform_sample<<<blocks, 256, 0, stream>>>(vf, of, awf, refpts, out, N, Lq);
  }
}
